// CueBasedRNNModel_6262062317722
// MI455X (gfx1250) — matrix operand order repaired and checked by reading, not compiled, not run on hardware
//
#include <hip/hip_runtime.h>
#include <hip/hip_bf16.h>

// ---------------- model constants ----------------
#define L_SEQ  256
#define BATCH  16
#define NIN    1024
#define NHID_  1024
#define NVOC   32001          // NTOK + 1
#define NTILV  2001           // ceil(32001/16)

typedef __bf16 v16bf __attribute__((ext_vector_type(16)));
typedef float  v8f   __attribute__((ext_vector_type(8)));

__device__ __forceinline__ unsigned short f2bf(float f) {
  union { float f; unsigned int u; } v; v.f = f;
  unsigned int u = v.u;
  return (unsigned short)((u + 0x7FFFu + ((u >> 16) & 1u)) >> 16);  // RNE
}

// ---------------- embedding gather: emb[l][b][h] = table[obs[l][b]][h] ----
__global__ void embed_k(const int* __restrict__ obs, const float* __restrict__ table,
                        float* __restrict__ emb) {
  int t = blockIdx.x * blockDim.x + threadIdx.x;        // over L*B*NIN
  int h  = t & (NIN - 1);
  int rb = t >> 10;                                     // l*16 + b
  int tok = obs[rb];
  emb[t] = table[(size_t)tok * NIN + h];
}

// ---------------- zero cache slot 0 for hid/k/v ----------------
__global__ void zero3_k(float* a, float* b, float* c, int n) {
  int t = blockIdx.x * blockDim.x + threadIdx.x;
  if (t < n) { a[t] = 0.f; b[t] = 0.f; c[t] = 0.f; }
}

// ---------------- pack W[K][N] f32 -> WMMA-B bf16 lane layout --------------
// B tile 32x16 per chunk: lanes 0-15 hold K=c*32+{0..7,16..23} (N=lane),
//                         lanes 16-31 hold K=c*32+{8..15,24..31} (N=lane-16).
__global__ void packW_k(const float* __restrict__ W, unsigned short* __restrict__ packed,
                        int chunks, int N, int Ntiles) {
  int t = blockIdx.x * blockDim.x + threadIdx.x;
  if (t >= Ntiles * chunks * 32) return;
  int lane = t & 31;
  int c    = (t >> 5) % chunks;
  int nt   = t / (32 * chunks);
  int n     = nt * 16 + (lane & 15);
  int kbase = c * 32 + ((lane < 16) ? 0 : 8);
  unsigned short __align__(16) vals[16];
#pragma unroll
  for (int e = 0; e < 16; ++e) {
    float x = (n < N) ? W[(size_t)(kbase + ((e < 8) ? e : e + 8)) * N + n] : 0.f;
    vals[e] = f2bf(x);
  }
  unsigned short* dst = packed + ((size_t)nt * chunks + c) * 512 + lane * 16;
  ((uint4*)dst)[0] = ((const uint4*)vals)[0];
  ((uint4*)dst)[1] = ((const uint4*)vals)[1];
}

// ---------------- pack concat(s0|s1|s2|s3) [16][K] -> WMMA-A bf16 layout ---
// A 16x32 per chunk: lane<16 row M=lane holds K {0..7,16..23};
//                    lane>=16 row M=lane-16 holds K {8..15,24..31}.
__global__ void pack_concat_k(const float* __restrict__ s0, const float* __restrict__ s1,
                              const float* __restrict__ s2, const float* __restrict__ s3,
                              unsigned short* __restrict__ packed, int K) {
  int t = blockIdx.x * blockDim.x + threadIdx.x;
  if (t >= (K >> 5) * 32) return;
  int c = t >> 5, lane = t & 31, m = lane & 15;
  unsigned short __align__(16) vals[16];
#pragma unroll
  for (int e = 0; e < 16; ++e) {
    int kk = (lane < 16) ? ((e < 8) ? e : e + 8) : ((e < 8) ? e + 8 : e + 16);
    int k  = c * 32 + kk;
    const float* s = (k < 1024) ? s0 : (k < 2048) ? s1 : (k < 3072) ? s2 : s3;
    vals[e] = f2bf(s[m * 1024 + (k & 1023)]);
  }
  unsigned short* dst = packed + (size_t)c * 512 + lane * 16;
  ((uint4*)dst)[0] = ((const uint4*)vals)[0];
  ((uint4*)dst)[1] = ((const uint4*)vals)[1];
}

// ---------------- pack src[M][K] rows -> WMMA-A layout (decode) ------------
__global__ void pack_rows_k(const float* __restrict__ src, unsigned short* __restrict__ packed,
                            int K, int chunks, int Mtiles) {
  int t = blockIdx.x * blockDim.x + threadIdx.x;
  if (t >= Mtiles * chunks * 32) return;
  int lane = t & 31;
  int c    = (t >> 5) % chunks;
  int mt   = t / (32 * chunks);
  int m = mt * 16 + (lane & 15);
  unsigned short __align__(16) vals[16];
#pragma unroll
  for (int e = 0; e < 16; ++e) {
    int kk = (lane < 16) ? ((e < 8) ? e : e + 8) : ((e < 8) ? e + 8 : e + 16);
    vals[e] = f2bf(src[(size_t)m * K + c * 32 + kk]);
  }
  unsigned short* dst = packed + ((size_t)mt * chunks + c) * 512 + lane * 16;
  ((uint4*)dst)[0] = ((const uint4*)vals)[0];
  ((uint4*)dst)[1] = ((const uint4*)vals)[1];
}

// ---------------- WMMA GEMM: one wave per 16x16 tile, K loop of 32 --------
__global__ void gemm_k(const unsigned short* __restrict__ pA, const unsigned short* __restrict__ pW,
                       float* __restrict__ out, int chunks, int Mtiles, int Ntiles,
                       int ldOut, const float* __restrict__ bias, int Nlimit) {
  int wpb  = blockDim.x >> 5;
  int tile = blockIdx.x * wpb + (threadIdx.x >> 5);
  if (tile >= Mtiles * Ntiles) return;
  int lane = threadIdx.x & 31;
  int mt = tile / Ntiles, nt = tile % Ntiles;
  const v16bf* A = (const v16bf*)(pA + (size_t)mt * chunks * 512);
  const v16bf* B = (const v16bf*)(pW + (size_t)nt * chunks * 512);
  v8f acc = {};
#pragma unroll 2
  for (int c = 0; c < chunks; ++c) {
    v16bf a = A[c * 32 + lane];
    v16bf b = B[c * 32 + lane];
    acc = __builtin_amdgcn_wmma_f32_16x16x32_bf16(false, a, false, b, (short)0, acc,
                                                  false, false);
  }
  int col = nt * 16 + (lane & 15);
  if (col < Nlimit) {
    int r0 = mt * 16 + ((lane < 16) ? 0 : 8);
    float bv = bias ? bias[col] : 0.f;
#pragma unroll
    for (int v = 0; v < 8; ++v)
      out[(size_t)(r0 + v) * ldOut + col] = acc[v] + bv;
  }
}

// ---------------- LayerNorm + tanh; optional f32 out, optional A-pack out --
__global__ void ln_tanh_k(const float* __restrict__ raw, const float* __restrict__ bias,
                          const float* __restrict__ g, const float* __restrict__ bb,
                          float* __restrict__ out, unsigned short* __restrict__ packed, int N) {
  int m = blockIdx.x, t = threadIdx.x;
  __shared__ float red[16];
  float s = 0.f, s2 = 0.f;
  for (int k = t; k < N; k += 256) {
    float x = raw[m * N + k] + bias[k];
    s += x; s2 += x * x;
  }
  for (int o = 16; o; o >>= 1) { s += __shfl_xor(s, o); s2 += __shfl_xor(s2, o); }
  int wave = t >> 5, lane = t & 31;
  if (lane == 0) { red[wave] = s; red[8 + wave] = s2; }
  __syncthreads();
  float ts = 0.f, ts2 = 0.f;
  for (int w = 0; w < 8; ++w) { ts += red[w]; ts2 += red[8 + w]; }
  float mu = ts / N;
  float rinv = rsqrtf(ts2 / N - mu * mu + 1e-5f);
  for (int k = t; k < N; k += 256) {
    float x = raw[m * N + k] + bias[k];
    float y = tanhf((x - mu) * rinv * g[k] + bb[k]);
    if (out) out[m * N + k] = y;
    if (packed) {   // scatter into WMMA-A bf16 layout
      int c = k >> 5, kk = k & 31, lane2, e;
      if (kk < 8)       { lane2 = m;      e = kk;      }
      else if (kk < 16) { lane2 = m + 16; e = kk - 8;  }
      else if (kk < 24) { lane2 = m;      e = kk - 8;  }
      else              { lane2 = m + 16; e = kk - 16; }
      packed[(size_t)c * 512 + lane2 * 16 + e] = f2bf(y);
    }
  }
}

// ---------------- LayerNorm + tanh on f[16][3072]; split into caches ------
__global__ void ln_split_k(const float* __restrict__ raw, const float* __restrict__ bias,
                           const float* __restrict__ g, const float* __restrict__ bb,
                           float* __restrict__ kdst, float* __restrict__ vdst,
                           float* __restrict__ hdst, float* __restrict__ hout) {
  const int N = 3072;
  int m = blockIdx.x, t = threadIdx.x;
  __shared__ float red[16];
  float s = 0.f, s2 = 0.f;
  for (int k = t; k < N; k += 256) {
    float x = raw[m * N + k] + bias[k];
    s += x; s2 += x * x;
  }
  for (int o = 16; o; o >>= 1) { s += __shfl_xor(s, o); s2 += __shfl_xor(s2, o); }
  int wave = t >> 5, lane = t & 31;
  if (lane == 0) { red[wave] = s; red[8 + wave] = s2; }
  __syncthreads();
  float ts = 0.f, ts2 = 0.f;
  for (int w = 0; w < 8; ++w) { ts += red[w]; ts2 += red[8 + w]; }
  float mu = ts / N;
  float rinv = rsqrtf(ts2 / N - mu * mu + 1e-5f);
  for (int k = t; k < N; k += 256) {
    float x = raw[m * N + k] + bias[k];
    float y = tanhf((x - mu) * rinv * g[k] + bb[k]);
    int seg = k >> 10, kr = k & 1023;
    float* d = (seg == 0) ? kdst : (seg == 1) ? vdst : hdst;
    d[m * 1024 + kr] = y;
    if (seg == 2) hout[m * 1024 + kr] = y;
  }
}

// ---------------- attention: scores over cache 0..i, softmax, weighted V --
__global__ void attn_k(const float* __restrict__ query, const float* __restrict__ k_c,
                       const float* __restrict__ v_c, const float* __restrict__ masks,
                       float* __restrict__ attn, int i) {
  int b = blockIdx.x, t = threadIdx.x;
  int wave = t >> 5, lane = t & 31;
  __shared__ float sc[L_SEQ + 1];
  __shared__ float redm[8], reds[8];
  const float* q = query + b * NHID_;
  for (int j = wave; j <= i; j += 8) {
    const float* kr = k_c + (size_t)(j * BATCH + b) * NHID_;
    float p = 0.f;
    for (int h = lane; h < NHID_; h += 32) p += kr[h] * q[h];
    for (int o = 16; o; o >>= 1) p += __shfl_xor(p, o);
    if (lane == 0)
      sc[j] = (p + masks[((size_t)b * L_SEQ + i) * L_SEQ + j]) * 0.03125f; // 1/sqrt(1024)
  }
  __syncthreads();
  float lm = -3.0e38f;
  for (int j = t; j <= i; j += 256) lm = fmaxf(lm, sc[j]);
  for (int o = 16; o; o >>= 1) lm = fmaxf(lm, __shfl_xor(lm, o));
  if (lane == 0) redm[wave] = lm;
  __syncthreads();
  lm = redm[0];
  for (int w = 1; w < 8; ++w) lm = fmaxf(lm, redm[w]);
  float ls = 0.f;
  for (int j = t; j <= i; j += 256) { float e = expf(sc[j] - lm); sc[j] = e; ls += e; }
  for (int o = 16; o; o >>= 1) ls += __shfl_xor(ls, o);
  if (lane == 0) reds[wave] = ls;
  __syncthreads();
  float tot = 0.f;
  for (int w = 0; w < 8; ++w) tot += reds[w];
  float inv = 1.f / tot;
  for (int h = t; h < NHID_; h += 256) {
    float a = 0.f;
    for (int j = 0; j <= i; ++j) a += sc[j] * v_c[(size_t)(j * BATCH + b) * NHID_ + h];
    attn[b * NHID_ + h] = a * inv;
  }
}

// =======================================================================
extern "C" void kernel_launch(void* const* d_in, const int* in_sizes, int n_in,
                              void* d_out, int out_size, void* d_ws, size_t ws_size,
                              hipStream_t stream) {
  (void)in_sizes; (void)n_in; (void)out_size; (void)ws_size;
  const int*   obs   = (const int*)  d_in[0];
  const float* masks = (const float*)d_in[1];
  const float* table = (const float*)d_in[2];
  const float* Wq   = (const float*)d_in[3];
  const float* bq   = (const float*)d_in[4];
  const float* g_qn = (const float*)d_in[5];
  const float* b_qn = (const float*)d_in[6];
  const float* Wi   = (const float*)d_in[7];
  const float* bi   = (const float*)d_in[8];
  const float* g_in = (const float*)d_in[9];
  const float* b_in = (const float*)d_in[10];
  const float* Wf   = (const float*)d_in[11];
  const float* bf_  = (const float*)d_in[12];
  const float* g_fn = (const float*)d_in[13];
  const float* b_fn = (const float*)d_in[14];
  const float* Wd   = (const float*)d_in[15];
  const float* bd   = (const float*)d_in[16];
  float* out = (float*)d_out;

  char* base = (char*)d_ws;
  size_t off = 0;
  auto alloc = [&](size_t bytes) -> void* {
    void* p = base + off;
    off += (bytes + 255) & ~(size_t)255;
    return p;
  };

  // packed bf16 weights (WMMA-B layout): Ntiles * chunks * 512 ushorts
  unsigned short* pWq = (unsigned short*)alloc((size_t)64  * 64  * 512 * 2);
  unsigned short* pWi = (unsigned short*)alloc((size_t)256 * 128 * 512 * 2);
  unsigned short* pWf = (unsigned short*)alloc((size_t)192 * 128 * 512 * 2);
  unsigned short* pWd = (unsigned short*)alloc((size_t)NTILV * 32 * 512 * 2);
  // activations / caches (f32)
  float* emb     = (float*)alloc((size_t)L_SEQ * BATCH * NIN * 4);
  float* hid_c   = (float*)alloc((size_t)(L_SEQ + 1) * BATCH * NHID_ * 4);
  float* k_c     = (float*)alloc((size_t)(L_SEQ + 1) * BATCH * NHID_ * 4);
  float* v_c     = (float*)alloc((size_t)(L_SEQ + 1) * BATCH * NHID_ * 4);
  float* hiddens = (float*)alloc((size_t)L_SEQ * BATCH * NHID_ * 4);
  unsigned short* pAdec = (unsigned short*)alloc((size_t)256 * 32 * 512 * 2);
  unsigned short* pAq = (unsigned short*)alloc((size_t)64  * 512 * 2);
  unsigned short* pAi = (unsigned short*)alloc((size_t)128 * 512 * 2);
  unsigned short* pAf = (unsigned short*)alloc((size_t)128 * 512 * 2);
  float* qraw  = (float*)alloc((size_t)16 * 1024 * 4);
  float* query = (float*)alloc((size_t)16 * 1024 * 4);
  float* attnb = (float*)alloc((size_t)16 * 1024 * 4);
  float* iraw  = (float*)alloc((size_t)16 * 4096 * 4);
  float* fraw  = (float*)alloc((size_t)16 * 3072 * 4);

  // ---- prep: embedding, cache slot 0 zeros, weight packing ----
  embed_k<<<(L_SEQ * BATCH * NIN) / 256, 256, 0, stream>>>(obs, table, emb);
  zero3_k<<<(BATCH * NHID_ + 255) / 256, 256, 0, stream>>>(hid_c, k_c, v_c, BATCH * NHID_);
  packW_k<<<(64  * 64  * 32 + 255) / 256, 256, 0, stream>>>(Wq, pWq, 64,  1024, 64);
  packW_k<<<(256 * 128 * 32 + 255) / 256, 256, 0, stream>>>(Wi, pWi, 128, 4096, 256);
  packW_k<<<(192 * 128 * 32 + 255) / 256, 256, 0, stream>>>(Wf, pWf, 128, 3072, 192);
  packW_k<<<(NTILV * 32 * 32 + 255) / 256, 256, 0, stream>>>(Wd, pWd, 32, NVOC, NTILV);

  // ---- recurrence: 256 sequential steps ----
  for (int i = 0; i < L_SEQ; ++i) {
    const float* e_i = emb   + (size_t)i * BATCH * NIN;
    const float* h_p = hid_c + (size_t)i * BATCH * NHID_;
    // query = tanh(LN([e, h_prev] @ Wq + bq))
    pack_concat_k<<<2048 / 256, 256, 0, stream>>>(e_i, h_p, e_i, e_i, pAq, 2048);
    gemm_k<<<16, 128, 0, stream>>>(pAq, pWq, qraw, 64, 1, 64, 1024, (const float*)nullptr, 1024);
    ln_tanh_k<<<16, 256, 0, stream>>>(qraw, bq, g_qn, b_qn, query,
                                      (unsigned short*)nullptr, 1024);
    // attention over cache 0..i
    attn_k<<<16, 256, 0, stream>>>(query, k_c, v_c, masks, attnb, i);
    // inter = tanh(LN([e, query, attn, h_prev] @ Wi + bi)); packed A for Wf fused in LN
    pack_concat_k<<<4096 / 256, 256, 0, stream>>>(e_i, query, attnb, h_p, pAi, 4096);
    gemm_k<<<64, 128, 0, stream>>>(pAi, pWi, iraw, 128, 1, 256, 4096, (const float*)nullptr, 4096);
    ln_tanh_k<<<16, 256, 0, stream>>>(iraw, bi, g_in, b_in, (float*)nullptr, pAf, 4096);
    // f = tanh(LN(inter @ Wf + bf)); split into k/v/h caches + hiddens
    gemm_k<<<48, 128, 0, stream>>>(pAf, pWf, fraw, 128, 1, 192, 3072, (const float*)nullptr, 3072);
    ln_split_k<<<16, 256, 0, stream>>>(fraw, bf_, g_fn, b_fn,
                                       k_c   + (size_t)(i + 1) * BATCH * NHID_,
                                       v_c   + (size_t)(i + 1) * BATCH * NHID_,
                                       hid_c + (size_t)(i + 1) * BATCH * NHID_,
                                       hiddens + (size_t)i * BATCH * NHID_);
  }

  // ---- decode: [4096 x 1024] @ [1024 x 32001] + bd ----
  pack_rows_k<<<(256 * 32 * 32) / 256, 256, 0, stream>>>(hiddens, pAdec, 1024, 32, 256);
  {
    int tiles = 256 * NTILV;
    gemm_k<<<(tiles + 3) / 4, 128, 0, stream>>>(pAdec, pWd, out, 32, 256, NTILV,
                                                NVOC, bd, NVOC);
  }
}
